// GCN_15736760172909
// MI455X (gfx1250) — matrix operand order repaired and checked by reading, not compiled, not run on hardware
//
#include <hip/hip_runtime.h>

// ---------------------------------------------------------------------------
// GCN 2-layer forward for MI455X (gfx1250, wave32).
//   xb,W1b,W2b = bf16 pack of inputs
//   h0b    = xb @ W1b              (bf16 WMMA, f32 accum, async-LDS staging)
//   h      = spmm(e0, h0b)         (bf16 gather, f32 atomic scatter-add)
//   hb     = bf16(relu(h + b1))
//   h2b    = hb @ W2b              (bf16 WMMA)
//   logits = spmm(e1, h2b) + b2
//   out    = log_softmax(logits)   (wave32 shuffle reduction)
// ---------------------------------------------------------------------------

typedef __attribute__((ext_vector_type(16))) __bf16 v16bf;
typedef __attribute__((ext_vector_type(8)))  float  v8f;

union Frag16 {
    unsigned int u[8];
    v16bf        v;
};

static __device__ __forceinline__ unsigned short f2bf(float f) {
    unsigned int u = __float_as_uint(f);
    unsigned int r = u + 0x7FFFu + ((u >> 16) & 1u);   // round-to-nearest-even
    return (unsigned short)(r >> 16);
}
static __device__ __forceinline__ unsigned int pack2(float lo, float hi) {
    return (unsigned int)f2bf(lo) | ((unsigned int)f2bf(hi) << 16);
}
static __device__ __forceinline__ float bf_lo(unsigned int u) {
    return __uint_as_float(u << 16);
}
static __device__ __forceinline__ float bf_hi(unsigned int u) {
    return __uint_as_float(u & 0xFFFF0000u);
}

// gfx1250 async global->LDS copy (ASYNCcnt tracked, bypasses VGPRs)
static __device__ __forceinline__ void async_copy_b128(unsigned lds_off,
                                                       unsigned long long gaddr) {
    asm volatile("global_load_async_to_lds_b128 %0, %1, off"
                 :: "v"(lds_off), "v"(gaddr) : "memory");
}
static __device__ __forceinline__ void wait_async0() {
    asm volatile("s_wait_asynccnt 0" ::: "memory");
}

// ---------------------------------------------------------------------------
// Tiled GEMM: C_bf16[M,N] = A_bf16[M,K] @ B_bf16[K,N], f32 WMMA accumulate.
// 64x64 tile, 8 waves/block, K-chunk 32, double-buffered async-LDS staging,
// one workgroup barrier per K-chunk.
// ---------------------------------------------------------------------------
#define A_DW 20   // A LDS row stride in dwords: 16 data + 4 pad (80B, 16B-aligned)
#define B_DW 36   // B LDS row stride in dwords: 32 data + 4 pad (144B, 16B-aligned)

template <int N, int K>
__global__ __launch_bounds__(256) void gemm_bf16_wmma(
    const unsigned short* __restrict__ A, const unsigned short* __restrict__ B,
    unsigned short* __restrict__ C, int M)
{
    __shared__ unsigned int lds_a[2][64 * A_DW];
    __shared__ unsigned int lds_b[2][32 * B_DW];

    const int t    = threadIdx.x;
    const int lane = t & 31;
    const int wave = t >> 5;
    const int m0   = blockIdx.x * 64;
    const int n0   = blockIdx.y * 64;
    const int sm   = (wave >> 2) * 2;   // wave owns subtile rows sm, sm+1
    const int sn   = wave & 3;          // subtile column

    // staging: A = 64 rows x 64B (4 x 16B segs); B = 32 rows x 128B (8 segs)
    const int ar = t >> 2, aseg = t & 3;
    const int br = t >> 3, bseg = t & 7;
    const bool aval = (m0 + ar) < M;

    const unsigned long long abase =
        (unsigned long long)(size_t)(A + (size_t)(m0 + ar) * K) + (unsigned)aseg * 16u;
    const unsigned long long bbase =
        (unsigned long long)(size_t)(B + (size_t)br * N + n0) + (unsigned)bseg * 16u;

    const unsigned aoff = (unsigned)(size_t)&lds_a[0][ar * A_DW + aseg * 4];
    const unsigned boff = (unsigned)(size_t)&lds_b[0][br * B_DW + bseg * 4];
    constexpr unsigned ABUF = 64 * A_DW * 4;   // bytes per A buffer
    constexpr unsigned BBUF = 32 * B_DW * 4;   // bytes per B buffer

    // prologue: stage chunk 0 into buffer 0
    if (aval) async_copy_b128(aoff, abase);
    async_copy_b128(boff, bbase);

    v8f acc0 = {};
    v8f acc1 = {};

    constexpr int NK = K / 32;
    for (int kc = 0; kc < NK; ++kc) {
        wait_async0();        // my writes into buf[kc&1] are in LDS
        __syncthreads();      // everyone's writes done; prev buffer reads done

        if (kc + 1 < NK) {    // stage next chunk into the other buffer
            const unsigned nb = (unsigned)((kc + 1) & 1);
            if (aval)
                async_copy_b128(aoff + nb * ABUF,
                                abase + (unsigned long long)(kc + 1) * 64u);
            async_copy_b128(boff + nb * BBUF,
                            bbase + (unsigned long long)(kc + 1) * 64u * (unsigned)N);
        }

        const unsigned int* la = lds_a[kc & 1];
        const unsigned int* lb = lds_b[kc & 1];

        Frag16 fa0, fa1, fb;
        const int lrow0 = (sm * 16 + (lane & 15)) * A_DW;
        const int lrow1 = lrow0 + 16 * A_DW;
        #pragma unroll
        for (int v = 0; v < 8; ++v) {
            // A 16x32 bf16 frag: k-pair dword = (v/4)*8 + (lane/16)*4 + (v%4)
            const int ki = ((v >> 2) << 3) + ((lane >> 4) << 2) + (v & 3);
            fa0.u[v] = la[lrow0 + ki];
            fa1.u[v] = la[lrow1 + ki];
            // B 32x16 bf16 frag: lane%16 indexes N, dword v holds k rows {2ki,2ki+1}
            fb.u[v]  = (unsigned int)((const unsigned short*)lb)[(2 * ki) * 2 * B_DW + (sn << 4) + (lane & 15)]
                     | ((unsigned int)((const unsigned short*)lb)[(2 * ki + 1) * 2 * B_DW + (sn << 4) + (lane & 15)] << 16);
        }

        acc0 = __builtin_amdgcn_wmma_f32_16x16x32_bf16(
            false, fa0.v, false, fb.v, (short)0, acc0, false, false);
        acc1 = __builtin_amdgcn_wmma_f32_16x16x32_bf16(
            false, fa1.v, false, fb.v, (short)0, acc1, false, false);
    }

    // ---- store C (bf16): M = v + 8*(lane/16), N = lane%16 within subtile ----
    const int col = n0 + sn * 16 + (lane & 15);
    const int r0  = m0 + sm * 16 + ((lane >> 4) << 3);
    unsigned short* p0 = C + (size_t)r0 * N + col;
    unsigned short* p1 = p0 + (size_t)16 * N;
    if (m0 + 64 <= M) {   // full tile fast path: immediate store offsets
        #pragma unroll
        for (int v = 0; v < 8; ++v) p0[v * N] = f2bf(acc0[v]);
        #pragma unroll
        for (int v = 0; v < 8; ++v) p1[v * N] = f2bf(acc1[v]);
    } else {
        #pragma unroll
        for (int v = 0; v < 8; ++v)
            if (r0 + v < M) p0[v * N] = f2bf(acc0[v]);
        #pragma unroll
        for (int v = 0; v < 8; ++v)
            if (r0 + 16 + v < M) p1[v * N] = f2bf(acc1[v]);
    }
}

// ---------------------------------------------------------------------------
// SpMM scatter: out[dst[e]] += w[e] * h[src[e]]; h is bf16, out is f32.
// FEAT/8 threads per edge; 16B gather -> 8 global_atomic_add_f32.
// ---------------------------------------------------------------------------
template <int FEAT>
__global__ __launch_bounds__(256) void spmm_atomic_bf16(
    const unsigned short* __restrict__ hb, const int* __restrict__ src,
    const int* __restrict__ dst, const float* __restrict__ w,
    float* __restrict__ out, int nEdges)
{
    constexpr int TPE = FEAT / 8;
    const long long gid = (long long)blockIdx.x * blockDim.x + threadIdx.x;
    const int e = (int)(gid / TPE);
    if (e >= nEdges) return;
    const int c = (int)(gid % TPE) * 8;

    const int   s  = src[e];
    const int   d  = dst[e];
    const float we = w[e];

    const uint4 u = *(const uint4*)(hb + (size_t)s * FEAT + c);
    float* o = out + (size_t)d * FEAT + c;
    atomicAdd(o + 0, bf_lo(u.x) * we);
    atomicAdd(o + 1, bf_hi(u.x) * we);
    atomicAdd(o + 2, bf_lo(u.y) * we);
    atomicAdd(o + 3, bf_hi(u.y) * we);
    atomicAdd(o + 4, bf_lo(u.z) * we);
    atomicAdd(o + 5, bf_hi(u.z) * we);
    atomicAdd(o + 6, bf_lo(u.w) * we);
    atomicAdd(o + 7, bf_hi(u.w) * we);
}

__global__ void fill_zero(float* __restrict__ p, long long n) {
    long long i = (long long)blockIdx.x * blockDim.x + threadIdx.x;
    const long long stride = (long long)gridDim.x * blockDim.x;
    for (; i < n; i += stride) p[i] = 0.0f;
}

// pack f32 -> bf16 pairs; n must be a multiple of 8
__global__ void f32_to_bf16_pack(const float* __restrict__ in,
                                 unsigned int* __restrict__ out, long long n) {
    const long long i = ((long long)blockIdx.x * blockDim.x + threadIdx.x) * 8;
    if (i >= n) return;
    const float4* p = (const float4*)(in + i);
    const float4 f0 = p[0], f1 = p[1];
    uint4 o;
    o.x = pack2(f0.x, f0.y);
    o.y = pack2(f0.z, f0.w);
    o.z = pack2(f1.x, f1.y);
    o.w = pack2(f1.z, f1.w);
    *(uint4*)(out + (i >> 1)) = o;
}

// hb = bf16(relu(h + b1)); 256 columns (power of two)
__global__ void bias_relu_bf16(const float* __restrict__ h,
                               const float* __restrict__ b,
                               unsigned int* __restrict__ hb, long long n) {
    const long long i = ((long long)blockIdx.x * blockDim.x + threadIdx.x) * 8;
    if (i >= n) return;
    const float4* p = (const float4*)(h + i);
    float4 f0 = p[0], f1 = p[1];
    const int c = (int)(i & 255);
    f0.x = fmaxf(f0.x + b[c + 0], 0.0f);
    f0.y = fmaxf(f0.y + b[c + 1], 0.0f);
    f0.z = fmaxf(f0.z + b[c + 2], 0.0f);
    f0.w = fmaxf(f0.w + b[c + 3], 0.0f);
    f1.x = fmaxf(f1.x + b[c + 4], 0.0f);
    f1.y = fmaxf(f1.y + b[c + 5], 0.0f);
    f1.z = fmaxf(f1.z + b[c + 6], 0.0f);
    f1.w = fmaxf(f1.w + b[c + 7], 0.0f);
    uint4 o;
    o.x = pack2(f0.x, f0.y);
    o.y = pack2(f0.z, f0.w);
    o.z = pack2(f1.x, f1.y);
    o.w = pack2(f1.z, f1.w);
    *(uint4*)(hb + (i >> 1)) = o;
}

// out[row,:] = log_softmax(logits[row,:] + b2), 64 classes, one wave32 per row
__global__ __launch_bounds__(256) void logsoftmax64(
    const float* __restrict__ logits, const float* __restrict__ b2,
    float* __restrict__ out, int rows)
{
    const int lane = threadIdx.x & 31;
    const int row  = blockIdx.x * 8 + (threadIdx.x >> 5);
    if (row >= rows) return;

    const float* in = logits + (size_t)row * 64;
    const float v0 = in[lane]      + b2[lane];
    const float v1 = in[lane + 32] + b2[lane + 32];

    float m = fmaxf(v0, v1);
    #pragma unroll
    for (int o = 16; o > 0; o >>= 1) m = fmaxf(m, __shfl_xor(m, o, 32));

    float s = __expf(v0 - m) + __expf(v1 - m);
    #pragma unroll
    for (int o = 16; o > 0; o >>= 1) s += __shfl_xor(s, o, 32);

    const float ls = m + __logf(s);
    float* o_ = out + (size_t)row * 64;
    o_[lane]      = v0 - ls;
    o_[lane + 32] = v1 - ls;
}

// ---------------------------------------------------------------------------
extern "C" void kernel_launch(void* const* d_in, const int* in_sizes, int n_in,
                              void* d_out, int out_size, void* d_ws, size_t ws_size,
                              hipStream_t stream) {
    (void)in_sizes; (void)n_in; (void)out_size; (void)ws_size;

    constexpr int N_NODES = 50000, N_EDGES = 800000;
    constexpr int NFEAT = 512, NHID = 256, NCLASS = 64;

    const float* x   = (const float*)d_in[0];
    const int* src0  = (const int*)d_in[1];
    const int* dst0  = (const int*)d_in[2];
    const float* w0  = (const float*)d_in[3];
    const int* src1  = (const int*)d_in[4];
    const int* dst1  = (const int*)d_in[5];
    const float* w1  = (const float*)d_in[6];
    const float* W1  = (const float*)d_in[7];
    const float* b1  = (const float*)d_in[8];
    const float* W2  = (const float*)d_in[9];
    const float* b2  = (const float*)d_in[10];
    float* out = (float*)d_out;

    // ---- workspace layout (~128.3 MB) ----
    char* ws = (char*)d_ws;
    // region0 @0 (51.2MB): xb; after gemm1 reused for hb / h2b / logits
    unsigned short* xb     = (unsigned short*)(ws);
    unsigned short* hb     = (unsigned short*)(ws);                 // reuse
    unsigned short* h2b    = (unsigned short*)(ws + 25600000);      // reuse
    float*          logits = (float*)        (ws + 32000000);      // reuse
    unsigned short* h0b    = (unsigned short*)(ws + 51200000);     // 25.6MB
    float*          h      = (float*)        (ws + 76800000);     // 51.2MB
    unsigned short* W1b    = (unsigned short*)(ws + 128000000);    // 256KB
    unsigned short* W2b    = (unsigned short*)(ws + 128262144);    // 32KB

    const int mTiles = (N_NODES + 63) / 64;

    // 0) pack inputs to bf16
    {
        const long long nx = (long long)N_NODES * NFEAT;
        f32_to_bf16_pack<<<(int)(nx / 8 / 256), 256, 0, stream>>>(x, (unsigned*)xb, nx);
        f32_to_bf16_pack<<<(NFEAT * NHID / 8 + 255) / 256, 256, 0, stream>>>(
            W1, (unsigned*)W1b, NFEAT * NHID);
        f32_to_bf16_pack<<<(NHID * NCLASS / 8 + 255) / 256, 256, 0, stream>>>(
            W2, (unsigned*)W2b, NHID * NCLASS);
    }

    // 1) h0b = xb @ W1b
    gemm_bf16_wmma<NHID, NFEAT><<<dim3(mTiles, NHID / 64), 256, 0, stream>>>(
        xb, W1b, h0b, N_NODES);

    // 2) h = 0; h[dst0] += w0 * h0b[src0]
    const long long nH = (long long)N_NODES * NHID;
    fill_zero<<<2048, 256, 0, stream>>>(h, nH);
    {
        const long long thr = (long long)N_EDGES * (NHID / 8);
        spmm_atomic_bf16<NHID><<<(int)((thr + 255) / 256), 256, 0, stream>>>(
            h0b, src0, dst0, w0, h, N_EDGES);
    }

    // 3) hb = bf16(relu(h + b1))
    bias_relu_bf16<<<(int)((nH / 8 + 255) / 256), 256, 0, stream>>>(
        h, b1, (unsigned*)hb, nH);

    // 4) h2b = hb @ W2b
    gemm_bf16_wmma<NCLASS, NHID><<<dim3(mTiles, NCLASS / 64), 256, 0, stream>>>(
        hb, W2b, h2b, N_NODES);

    // 5) logits = 0; logits[dst1] += w1 * h2b[src1]
    const long long nL = (long long)N_NODES * NCLASS;
    fill_zero<<<2048, 256, 0, stream>>>(logits, nL);
    {
        const long long thr = (long long)N_EDGES * (NCLASS / 8);
        spmm_atomic_bf16<NCLASS><<<(int)((thr + 255) / 256), 256, 0, stream>>>(
            h2b, src1, dst1, w1, logits, N_EDGES);
    }

    // 6) out = log_softmax(logits + b2)
    logsoftmax64<<<(N_NODES + 7) / 8, 256, 0, stream>>>(logits, b2, out, N_NODES);
}
